// SPROUT_BrainLike_68341519614268
// MI455X (gfx1250) — matrix operand order repaired and checked by reading, not compiled, not run on hardware
//
#include <hip/hip_runtime.h>
#include <cmath>

// =====================================================================
// SPROUT brain-like sparse network on MI455X (gfx1250).
// FP16 WMMA (v_wmma_f32_16x16x32_f16) for every GEMM, FP32 accumulate,
// fused bias/activation/scale epilogues, double-buffered LDS 64x64 block
// tiles (8 wave32 waves), branchless clamped staging, batched strides
// for attention, global_prefetch_b8 for the K+2 slab.
// =====================================================================

typedef _Float16 f16;
typedef __attribute__((ext_vector_type(16))) _Float16 v16h;
typedef __attribute__((ext_vector_type(8)))  float    v8f;

#define BSZ  32
#define SEQL 512
#define DM   256
#define NHD  4
#define HDIM 64
#define NN   4096
#define VOC  30000

#define TBM 64
#define TBN 64
#define TBK 32
#define LDP (TBK + 8)

// ---------------------------------------------------------------------
// helpers
// ---------------------------------------------------------------------
__device__ inline float wave_sum(float v) {
  #pragma unroll
  for (int m = 16; m > 0; m >>= 1) v += __shfl_xor(v, m, 32);
  return v;
}
__device__ inline float wave_max(float v) {
  #pragma unroll
  for (int m = 16; m > 0; m >>= 1) v = fmaxf(v, __shfl_xor(v, m, 32));
  return v;
}

template<int ACT>
__device__ inline float apply_act(float x) {
  if constexpr (ACT == 1) return fmaxf(x, 0.f);                                    // relu
  else if constexpr (ACT == 2) return 0.5f * x * (1.f + erff(x * 0.70710678118f)); // gelu exact
  else return x;
}

// ---------------------------------------------------------------------
// WMMA GEMM:  C[M,N] = act( alpha * A[M,K] x B^T + bias )
//   A row-major f16 [M,K].
//   B_NK=true : B stored [N,K] (weight layout, out x in)
//   B_NK=false: B stored [K,N] (e.g. V matrix in attention)
//   batched via gridDim.z with element strides sA/sB/sC.
// Staging is branchless: OOB rows/cols are clamped; their products only
// land in C rows/cols that the guarded epilogue never stores.
// ---------------------------------------------------------------------
template<int ACT, bool B_NK, typename OUT>
__global__ __launch_bounds__(256)
void gemm_wmma(const f16* __restrict__ A, const f16* __restrict__ Bm,
               const float* __restrict__ bias, OUT* __restrict__ C,
               int M, int N, int K, long sA, long sB, long sC, float alpha) {
  __shared__ f16 As[2][TBM][LDP];
  __shared__ f16 Bs[2][TBN][LDP];   // stored [n][k]

  A  += (long)blockIdx.z * sA;
  Bm += (long)blockIdx.z * sB;
  C  += (long)blockIdx.z * sC;

  const int tid  = threadIdx.x;
  const int wid  = tid >> 5;
  const int lane = tid & 31;
  const int m0 = blockIdx.y * TBM;
  const int n0 = blockIdx.x * TBN;
  const int wm = (wid >> 1) * 16;   // wave row offset within tile
  const int wn = (wid & 1) * 32;    // wave col offset within tile
  const int row = lane & 15;
  const int kh  = lane >> 4;

  v8f acc0 = {0.f, 0.f, 0.f, 0.f, 0.f, 0.f, 0.f, 0.f};
  v8f acc1 = {0.f, 0.f, 0.f, 0.f, 0.f, 0.f, 0.f, 0.f};

  const int lr = tid >> 2;          // 0..63 row within tile
  const int lc = (tid & 3) * 8;     // 0,8,16,24 col

  const int ga = (m0 + lr < M) ? (m0 + lr) : (M - 1);
  const int gb = (n0 + lr < N) ? (n0 + lr) : (N - 1);
  const f16* aptr = A + (long)ga * K + lc;
  const f16* bptr = B_NK ? (Bm + (long)gb * K + lc) : (Bm + gb);

  uint4 ra, rb;
  f16 rbk[8];

  // ---- prologue: stage K-slab 0 into buffer 0 ----
  ra = *reinterpret_cast<const uint4*>(aptr);
  if constexpr (B_NK) {
    rb = *reinterpret_cast<const uint4*>(bptr);
  } else {
    #pragma unroll
    for (int j = 0; j < 8; ++j) rbk[j] = bptr[(long)(lc + j) * N];
  }
  *reinterpret_cast<uint4*>(&As[0][lr][lc]) = ra;
  if constexpr (B_NK) {
    *reinterpret_cast<uint4*>(&Bs[0][lr][lc]) = rb;
  } else {
    #pragma unroll
    for (int j = 0; j < 8; ++j) Bs[0][lr][lc + j] = rbk[j];
  }
  __syncthreads();

  int cur = 0;
  for (int k0 = 0; k0 < K; k0 += TBK) {
    const bool has_next = (k0 + TBK) < K;

    // issue next slab's global loads before the math (latency overlap)
    if (has_next) {
      ra = *reinterpret_cast<const uint4*>(aptr + k0 + TBK);
      if constexpr (B_NK) {
        rb = *reinterpret_cast<const uint4*>(bptr + k0 + TBK);
        if (k0 + 2 * TBK < K) {
          __builtin_prefetch(aptr + k0 + 2 * TBK, 0, 0);
          __builtin_prefetch(bptr + k0 + 2 * TBK, 0, 0);
        }
      } else {
        #pragma unroll
        for (int j = 0; j < 8; ++j) rbk[j] = bptr[(long)(k0 + TBK + lc + j) * N];
      }
    }

    // ---- fragments: contiguous k = 16*kh + s in A and B ----
    v16h af, bf0, bf1;
    #pragma unroll
    for (int v = 0; v < 8; ++v) {
      int kb = 2 * v + 16 * kh;
      af[2 * v]     = As[cur][wm + row][kb];
      af[2 * v + 1] = As[cur][wm + row][kb + 1];
    }
    #pragma unroll
    for (int v = 0; v < 8; ++v) {
      int kb = 2 * v + 16 * kh;
      bf0[2 * v]     = Bs[cur][wn + row][kb];
      bf0[2 * v + 1] = Bs[cur][wn + row][kb + 1];
      bf1[2 * v]     = Bs[cur][wn + 16 + row][kb];
      bf1[2 * v + 1] = Bs[cur][wn + 16 + row][kb + 1];
    }
    acc0 = __builtin_amdgcn_wmma_f32_16x16x32_f16(false, af, false, bf0,
                                                  (short)0, acc0, false, false);
    acc1 = __builtin_amdgcn_wmma_f32_16x16x32_f16(false, af, false, bf1,
                                                  (short)0, acc1, false, false);

    if (has_next) {
      const int nxt = cur ^ 1;
      *reinterpret_cast<uint4*>(&As[nxt][lr][lc]) = ra;
      if constexpr (B_NK) {
        *reinterpret_cast<uint4*>(&Bs[nxt][lr][lc]) = rb;
      } else {
        #pragma unroll
        for (int j = 0; j < 8; ++j) Bs[nxt][lr][lc + j] = rbk[j];
      }
      __syncthreads();
      cur = nxt;
    }
  }

  // ---- epilogue: C layout lane=(n), vgpr v -> m = v + 8*kh ----
  const int nl = lane & 15;
  #pragma unroll
  for (int v = 0; v < 8; ++v) {
    int cm = m0 + wm + v + 8 * kh;
    if (cm >= M) continue;
    long base = (long)cm * N;
    int c0 = n0 + wn + nl;
    if (c0 < N) {
      float r = acc0[v] * alpha + (bias ? bias[c0] : 0.f);
      C[base + c0] = (OUT)apply_act<ACT>(r);
    }
    int c1 = c0 + 16;
    if (c1 < N) {
      float r = acc1[v] * alpha + (bias ? bias[c1] : 0.f);
      C[base + c1] = (OUT)apply_act<ACT>(r);
    }
  }
}

// ---------------------------------------------------------------------
// elementwise / reduction kernels
// ---------------------------------------------------------------------
__global__ void cvt_f32_f16_kernel(const float* __restrict__ in, f16* __restrict__ out, long n) {
  long i = (long)blockIdx.x * blockDim.x + threadIdx.x;
  long stride = (long)gridDim.x * blockDim.x;
  for (; i < n; i += stride) out[i] = (f16)in[i];
}

__global__ void fill_zero_f32_kernel(float* __restrict__ p, long n) {
  long i = (long)blockIdx.x * blockDim.x + threadIdx.x;
  long stride = (long)gridDim.x * blockDim.x;
  for (; i < n; i += stride) p[i] = 0.f;
}

__global__ void embed_kernel(const int* __restrict__ tokens,
                             const float* __restrict__ tok_emb,
                             const float* __restrict__ pos_emb,
                             float* __restrict__ xf, f16* __restrict__ xh) {
  long i = (long)blockIdx.x * blockDim.x + threadIdx.x;
  long total = (long)BSZ * SEQL * DM;
  if (i >= total) return;
  int d = (int)(i % DM);
  long bs = i / DM;
  int s = (int)(bs % SEQL);
  int t = tokens[bs];
  float v = tok_emb[(long)t * DM + d] + pos_emb[(long)s * DM + d];
  xf[i] = v;
  xh[i] = (f16)v;
}

// LayerNorm over D=256, optional residual; wave per row; optional f32/f16 outs.
__global__ __launch_bounds__(256)
void add_ln_kernel(const float* __restrict__ x, const float* __restrict__ res,
                   const float* __restrict__ g, const float* __restrict__ b,
                   float* __restrict__ of, f16* __restrict__ oh, long rows) {
  long row = (long)blockIdx.x * 8 + (threadIdx.x >> 5);
  int lane = threadIdx.x & 31;
  if (row >= rows) return;
  const float* xr = x + row * DM;
  const float* rr = res ? res + row * DM : nullptr;
  float v[8];
  float s = 0.f;
  #pragma unroll
  for (int j = 0; j < 8; ++j) {
    int c = lane + j * 32;
    float t = xr[c] + (rr ? rr[c] : 0.f);
    v[j] = t; s += t;
  }
  float mu = wave_sum(s) * (1.f / DM);
  float vs = 0.f;
  #pragma unroll
  for (int j = 0; j < 8; ++j) { float d = v[j] - mu; vs += d * d; }
  float var = wave_sum(vs) * (1.f / DM);
  float rstd = rsqrtf(var + 1e-5f);
  #pragma unroll
  for (int j = 0; j < 8; ++j) {
    int c = lane + j * 32;
    float o = (v[j] - mu) * rstd * g[c] + b[c];
    if (of) of[row * DM + c] = o;
    if (oh) oh[row * DM + c] = (f16)o;
  }
}

// softmax over rows (wave per row), writes f16 probs
__global__ __launch_bounds__(256)
void softmax_kernel(const float* __restrict__ in, f16* __restrict__ out,
                    long rows, int cols) {
  long row = (long)blockIdx.x * 8 + (threadIdx.x >> 5);
  int lane = threadIdx.x & 31;
  if (row >= rows) return;
  const float* r = in + row * cols;
  float m = -3.0e38f;
  for (int c = lane; c < cols; c += 32) m = fmaxf(m, r[c]);
  m = wave_max(m);
  float s = 0.f;
  for (int c = lane; c < cols; c += 32) s += __expf(r[c] - m);
  s = wave_sum(s);
  float inv = 1.f / s;
  for (int c = lane; c < cols; c += 32)
    out[row * cols + c] = (f16)(__expf(r[c] - m) * inv);
}

// qkv [B,S,3D] f16  ->  Qh/Kh/Vh [B*H, S, HD] f16
__global__ void split_qkv_kernel(const f16* __restrict__ qkv,
                                 f16* __restrict__ Qh, f16* __restrict__ Kh,
                                 f16* __restrict__ Vh, int Bn, int Sn) {
  long i = (long)blockIdx.x * blockDim.x + threadIdx.x;
  long total = (long)Bn * Sn * 3 * DM;
  if (i >= total) return;
  int col = (int)(i % (3 * DM));
  long bs = i / (3 * DM);
  int s = (int)(bs % Sn);
  int b = (int)(bs / Sn);
  int part = col / DM;
  int d = col % DM;
  int h = d / HDIM, e = d % HDIM;
  long dst = ((long)(b * NHD + h) * Sn + s) * HDIM + e;
  f16 v = qkv[i];
  if (part == 0) Qh[dst] = v; else if (part == 1) Kh[dst] = v; else Vh[dst] = v;
}

// Oh [B*H, S, HD] f16 -> out [B,S,D] f16
__global__ void merge_heads_kernel(const f16* __restrict__ Oh, f16* __restrict__ out,
                                   int Bn, int Sn) {
  long i = (long)blockIdx.x * blockDim.x + threadIdx.x;
  long total = (long)Bn * Sn * DM;
  if (i >= total) return;
  int d = (int)(i % DM);
  long bs = i / DM;
  int s = (int)(bs % Sn);
  int b = (int)(bs / Sn);
  int h = d / HDIM, e = d % HDIM;
  out[i] = Oh[((long)(b * NHD + h) * Sn + s) * HDIM + e];
}

// iterative top-k, one block per row (n=4096, k<=256)
__global__ __launch_bounds__(256)
void topk_kernel(const float* __restrict__ x, int n, int k,
                 int* __restrict__ idx_out, float* __restrict__ val_out) {
  __shared__ float sv[NN];
  __shared__ float rmax[256];
  __shared__ int   rarg[256];
  const float* row = x + (long)blockIdx.x * n;
  for (int i = threadIdx.x; i < n; i += 256) sv[i] = row[i];
  __syncthreads();
  for (int it = 0; it < k; ++it) {
    float best = -3.0e38f; int barg = 0;
    for (int i = threadIdx.x; i < n; i += 256)
      if (sv[i] > best) { best = sv[i]; barg = i; }
    rmax[threadIdx.x] = best; rarg[threadIdx.x] = barg;
    __syncthreads();
    for (int s = 128; s > 0; s >>= 1) {
      if (threadIdx.x < s && rmax[threadIdx.x + s] > rmax[threadIdx.x]) {
        rmax[threadIdx.x] = rmax[threadIdx.x + s];
        rarg[threadIdx.x] = rarg[threadIdx.x + s];
      }
      __syncthreads();
    }
    if (threadIdx.x == 0) {
      idx_out[(long)blockIdx.x * k + it] = rarg[0];
      val_out[(long)blockIdx.x * k + it] = rmax[0];
      sv[rarg[0]] = -3.0e38f;
    }
    __syncthreads();
  }
}

__global__ void scatter_vals_kernel(const int* __restrict__ idx, const float* __restrict__ vals,
                                    float* __restrict__ act, int n, int k, int do_sigmoid) {
  long i = (long)blockIdx.x * blockDim.x + threadIdx.x;
  if (i >= (long)BSZ * k) return;
  int b = (int)(i / k);
  float v = vals[i];
  if (do_sigmoid) v = 1.f / (1.f + __expf(-v));
  act[(long)b * n + idx[i]] = v;
}

__global__ void init_hidden_kernel(const float* __restrict__ sig, float* __restrict__ hidden) {
  long i = (long)blockIdx.x * blockDim.x + threadIdx.x;
  long total = (long)BSZ * NN * DM;
  if (i >= total) return;
  long nd = i % ((long)NN * DM);
  hidden[i] = sig[nd];
}

__global__ void gather_rows_kernel(const float* __restrict__ hidden, const int* __restrict__ idx,
                                   f16* __restrict__ out, int k) {
  long i = (long)blockIdx.x * blockDim.x + threadIdx.x;
  long total = (long)BSZ * k * DM;
  if (i >= total) return;
  int d = (int)(i % DM);
  long t = i / DM;
  int j = (int)(t % k), b = (int)(t / k);
  long src = ((long)b * NN + idx[(long)b * k + j]) * DM + d;
  out[i] = (f16)hidden[src];
}

// comb[:, :256]=f16(hidden); comb[:, 256:]=0
__global__ void build_comb_kernel(const float* __restrict__ hidden, f16* __restrict__ comb) {
  long i = (long)blockIdx.x * blockDim.x + threadIdx.x;
  long total = (long)BSZ * NN * DM;
  if (i >= total) return;
  int d = (int)(i % DM);
  long r = i / DM;
  comb[r * (2 * DM) + d] = (f16)hidden[i];
  comb[r * (2 * DM) + DM + d] = (f16)0.f;
}

__global__ void scatter_msg_kernel(f16* __restrict__ comb, const int* __restrict__ idx,
                                   const float* __restrict__ vals, const float* __restrict__ msg,
                                   int k) {
  long i = (long)blockIdx.x * blockDim.x + threadIdx.x;
  long total = (long)BSZ * k * DM;
  if (i >= total) return;
  int d = (int)(i % DM);
  long t = i / DM;
  int j = (int)(t % k), b = (int)(t / k);
  long r = (long)b * NN + idx[(long)b * k + j];
  comb[r * (2 * DM) + DM + d] = (f16)(msg[i] * vals[(long)b * k + j]);
}

// comb[:, :256]=f16(hidden); comb[:, 256:]=new_hidden (already f16)
__global__ void build_comb2_kernel(const float* __restrict__ hidden, const f16* __restrict__ nhh,
                                   f16* __restrict__ comb) {
  long i = (long)blockIdx.x * blockDim.x + threadIdx.x;
  long total = (long)BSZ * NN * DM;
  if (i >= total) return;
  int d = (int)(i % DM);
  long r = i / DM;
  comb[r * (2 * DM) + d] = (f16)hidden[i];
  comb[r * (2 * DM) + DM + d] = nhh[i];
}

// delta[r] = sigmoid( dot(w, x[r]) + b0 ), wave per row
__global__ __launch_bounds__(256)
void rowdot_sigmoid_kernel(const f16* __restrict__ x, const float* __restrict__ w,
                           const float* __restrict__ b, float* __restrict__ out, long rows) {
  long row = (long)blockIdx.x * 8 + (threadIdx.x >> 5);
  int lane = threadIdx.x & 31;
  if (row >= rows) return;
  float s = 0.f;
  for (int c = lane; c < DM; c += 32) s += (float)x[row * DM + c] * w[c];
  s = wave_sum(s);
  if (lane == 0) out[row] = 1.f / (1.f + __expf(-(s + b[0])));
}

__global__ void update_act_kernel(const float* __restrict__ act, const float* __restrict__ delta,
                                  float* __restrict__ out) {
  long i = (long)blockIdx.x * blockDim.x + threadIdx.x;
  if (i >= (long)BSZ * NN) return;
  float v = 0.7f * act[i] + 0.3f * delta[i];
  out[i] = fminf(fmaxf(v, 0.f), 1.f);
}

__global__ void scatter_rows_kernel(const float* __restrict__ nh, const int* __restrict__ idx,
                                    float* __restrict__ hnext, int k) {
  long i = (long)blockIdx.x * blockDim.x + threadIdx.x;
  long total = (long)BSZ * k * DM;
  if (i >= total) return;
  int d = (int)(i % DM);
  long t = i / DM;
  int j = (int)(t % k), b = (int)(t / k);
  long r = ((long)b * NN + idx[(long)b * k + j]) * DM + d;
  hnext[r] = nh[r];
}

__global__ void bcast_row_kernel(const float* __restrict__ q, f16* __restrict__ out) {
  long i = (long)blockIdx.x * blockDim.x + threadIdx.x;
  if (i >= (long)BSZ * DM) return;
  out[i] = (f16)q[i % DM];
}

// ---------------------------------------------------------------------
// host side
// ---------------------------------------------------------------------
struct Bump {
  char* base; size_t off; size_t cap;
  void* take(size_t bytes) {
    size_t a = (off + 255) & ~(size_t)255;
    off = a + bytes;
    return base + a;
  }
};
template<typename T> static inline T* balloc(Bump& b, long n) {
  return (T*)b.take((size_t)n * sizeof(T));
}

static inline int gblocks(long n) { return (int)((n + 255) / 256); }
static inline int rblocks(long rows) { return (int)((rows + 7) / 8); }

static inline void cvt(hipStream_t st, const float* s, f16* d, long n) {
  long b = (n + 255) / 256; if (b > 65535) b = 65535;
  cvt_f32_f16_kernel<<<(int)b, 256, 0, st>>>(s, d, n);
}
static inline void zfill(hipStream_t st, float* p, long n) {
  long b = (n + 255) / 256; if (b > 65535) b = 65535;
  fill_zero_f32_kernel<<<(int)b, 256, 0, st>>>(p, n);
}

template<int ACT, bool BNK, typename OUT>
static inline void gemm(hipStream_t st, const f16* A, const f16* Bm, const float* bias,
                        OUT* C, int M, int N, int K, int batch = 1,
                        long sA = 0, long sB = 0, long sC = 0, float alpha = 1.f) {
  dim3 grid((N + TBN - 1) / TBN, (M + TBM - 1) / TBM, batch);
  hipLaunchKernelGGL((gemm_wmma<ACT, BNK, OUT>), grid, dim3(256), 0, st,
                     A, Bm, bias, C, M, N, K, sA, sB, sC, alpha);
}

#define FP(i) ((const float*)d_in[(i)])

extern "C" void kernel_launch(void* const* d_in, const int* in_sizes, int n_in,
                              void* d_out, int out_size, void* d_ws, size_t ws_size,
                              hipStream_t stream) {
  (void)in_sizes; (void)n_in; (void)out_size;
  hipStream_t st = stream;
  Bump ws{(char*)d_ws, 0, ws_size};

  const int* tokens = (const int*)d_in[0];

  // ---- weight conversion to f16 ----
  f16 *enc_wi[2], *enc_wo[2], *enc_l1[2], *enc_l2[2];
  for (int l = 0; l < 2; ++l) {
    int LB = 3 + 12 * l;
    enc_wi[l] = balloc<f16>(ws, 768 * 256);  cvt(st, FP(LB + 0), enc_wi[l], 768 * 256);
    enc_wo[l] = balloc<f16>(ws, 256 * 256);  cvt(st, FP(LB + 2), enc_wo[l], 256 * 256);
    enc_l1[l] = balloc<f16>(ws, 1024 * 256); cvt(st, FP(LB + 6), enc_l1[l], 1024 * 256);
    enc_l2[l] = balloc<f16>(ws, 256 * 1024); cvt(st, FP(LB + 8), enc_l2[l], 256 * 1024);
  }
  f16* pa_wi = balloc<f16>(ws, 768 * 256);        cvt(st, FP(28), pa_wi, 768 * 256);
  f16* pa_wo = balloc<f16>(ws, 256 * 256);        cvt(st, FP(30), pa_wo, 256 * 256);
  f16* act1w = balloc<f16>(ws, 512 * 256);        cvt(st, FP(32), act1w, 512 * 256);
  f16* act2w = balloc<f16>(ws, (long)NN * 512);   cvt(st, FP(34), act2w, (long)NN * 512);
  f16* ma_wi = balloc<f16>(ws, 768 * 256);        cvt(st, FP(37), ma_wi, 768 * 256);
  f16* ma_wo = balloc<f16>(ws, 256 * 256);        cvt(st, FP(39), ma_wo, 256 * 256);
  f16* su1w  = balloc<f16>(ws, 256 * 512);        cvt(st, FP(41), su1w, 256 * 512);
  f16* su2w  = balloc<f16>(ws, 256 * 256);        cvt(st, FP(43), su2w, 256 * 256);
  f16* au1w  = balloc<f16>(ws, 256 * 512);        cvt(st, FP(47), au1w, 256 * 512);
  f16* dec1w = balloc<f16>(ws, (long)512 * NN);   cvt(st, FP(51), dec1w, (long)512 * NN);
  f16* dec2w = balloc<f16>(ws, 256 * 512);        cvt(st, FP(53), dec2w, 256 * 512);
  f16* logw  = balloc<f16>(ws, (long)VOC * 256);  cvt(st, FP(57), logw, (long)VOC * 256);

  // ---- activations / scratch ----
  const long BS   = (long)BSZ * SEQL;        // 16384
  const long BND  = (long)BSZ * NN * DM;     // 33.55M
  float* xf       = balloc<float>(ws, BS * DM);
  f16*   xh       = balloc<f16>(ws, BS * DM);
  float* tmpO     = balloc<float>(ws, BS * DM);            // attn out / ffn2 out
  f16*   ffn1h    = balloc<f16>(ws, BS * 4 * DM);
  f16*   qkvh     = balloc<f16>(ws, BS * 3 * DM);
  f16*   Qh       = balloc<f16>(ws, BS * DM);
  f16*   Kh       = balloc<f16>(ws, BS * DM);
  f16*   Vh       = balloc<f16>(ws, BS * DM);
  float* scores   = balloc<float>(ws, (long)BSZ * NHD * SEQL * SEQL);
  f16*   probs    = balloc<f16>(ws, (long)BSZ * NHD * SEQL * SEQL);
  f16*   Oh       = balloc<f16>(ws, BS * DM);
  f16*   ctx      = balloc<f16>(ws, BS * DM);
  f16*   poolqh   = balloc<f16>(ws, (long)BSZ * DM);
  f16*   qph      = balloc<f16>(ws, (long)BSZ * DM);
  float* pooled   = balloc<float>(ws, (long)BSZ * DM);
  f16*   pooledh  = balloc<f16>(ws, (long)BSZ * DM);
  f16*   a1h      = balloc<f16>(ws, (long)BSZ * 512);
  float* logits0  = balloc<float>(ws, (long)BSZ * NN);
  float* acta     = balloc<float>(ws, (long)BSZ * NN);
  float* actb     = balloc<float>(ws, (long)BSZ * NN);
  float* delta    = balloc<float>(ws, (long)BSZ * NN);
  int*   idx      = balloc<int>(ws, (long)BSZ * 256);
  float* vals     = balloc<float>(ws, (long)BSZ * 256);
  float* hid0     = balloc<float>(ws, BND);
  float* hid1     = balloc<float>(ws, BND);
  f16*   comb     = balloc<f16>(ws, BND * 2);
  f16*   t1h      = balloc<f16>(ws, BND);
  float* nh       = balloc<float>(ws, BND);
  f16*   nhh      = balloc<f16>(ws, BND);
  f16*   activeh  = balloc<f16>(ws, (long)BSZ * 256 * DM);
  float* msgout   = balloc<float>(ws, (long)BSZ * 256 * DM);
  f16*   acth     = balloc<f16>(ws, (long)BSZ * NN);
  f16*   d1h      = balloc<f16>(ws, (long)BSZ * 512);
  float* d2       = balloc<float>(ws, (long)BSZ * DM);
  f16*   dnh      = balloc<f16>(ws, (long)BSZ * DM);

  // ================= embedding =================
  embed_kernel<<<gblocks(BS * DM), 256, 0, st>>>(tokens, FP(1), FP(2), xf, xh);

  // ================= transformer encoder (2 layers) =================
  for (int l = 0; l < 2; ++l) {
    int LB = 3 + 12 * l;
    gemm<0, true, f16>(st, xh, enc_wi[l], FP(LB + 1), qkvh, (int)BS, 3 * DM, DM);
    split_qkv_kernel<<<gblocks(BS * 3 * DM), 256, 0, st>>>(qkvh, Qh, Kh, Vh, BSZ, SEQL);
    gemm<0, true, float>(st, Qh, Kh, nullptr, scores, SEQL, SEQL, HDIM,
                         BSZ * NHD, (long)SEQL * HDIM, (long)SEQL * HDIM,
                         (long)SEQL * SEQL, 0.125f);
    softmax_kernel<<<rblocks((long)BSZ * NHD * SEQL), 256, 0, st>>>(
        scores, probs, (long)BSZ * NHD * SEQL, SEQL);
    gemm<0, false, f16>(st, probs, Vh, nullptr, Oh, SEQL, HDIM, SEQL,
                        BSZ * NHD, (long)SEQL * SEQL, (long)SEQL * HDIM,
                        (long)SEQL * HDIM, 1.f);
    merge_heads_kernel<<<gblocks(BS * DM), 256, 0, st>>>(Oh, ctx, BSZ, SEQL);
    gemm<0, true, float>(st, ctx, enc_wo[l], FP(LB + 3), tmpO, (int)BS, DM, DM);
    add_ln_kernel<<<rblocks(BS), 256, 0, st>>>(xf, tmpO, FP(LB + 4), FP(LB + 5), xf, xh, BS);
    gemm<1, true, f16>(st, xh, enc_l1[l], FP(LB + 7), ffn1h, (int)BS, 4 * DM, DM);
    gemm<0, true, float>(st, ffn1h, enc_l2[l], FP(LB + 9), tmpO, (int)BS, DM, 4 * DM);
    add_ln_kernel<<<rblocks(BS), 256, 0, st>>>(xf, tmpO, FP(LB + 10), FP(LB + 11), xf, xh, BS);
  }

  // ================= attention pooling =================
  gemm<0, true, f16>(st, xh, pa_wi, FP(29), qkvh, (int)BS, 3 * DM, DM);
  split_qkv_kernel<<<gblocks(BS * 3 * DM), 256, 0, st>>>(qkvh, Qh, Kh, Vh, BSZ, SEQL);
  bcast_row_kernel<<<gblocks((long)BSZ * DM), 256, 0, st>>>(FP(27), poolqh);
  gemm<0, true, f16>(st, poolqh, pa_wi, FP(29), qph, BSZ, DM, DM); // q-proj = first D rows of wi
  gemm<0, true, float>(st, qph, Kh, nullptr, scores, 1, SEQL, HDIM,
                       BSZ * NHD, (long)HDIM, (long)SEQL * HDIM, (long)SEQL, 0.125f);
  softmax_kernel<<<rblocks(BSZ * NHD), 256, 0, st>>>(scores, probs, BSZ * NHD, SEQL);
  gemm<0, false, f16>(st, probs, Vh, nullptr, Oh, 1, HDIM, SEQL,
                      BSZ * NHD, (long)SEQL, (long)SEQL * HDIM, (long)HDIM, 1.f);
  merge_heads_kernel<<<gblocks((long)BSZ * DM), 256, 0, st>>>(Oh, ctx, BSZ, 1);
  gemm<0, true, float>(st, ctx, pa_wo, FP(31), pooled, BSZ, DM, DM);
  cvt(st, pooled, pooledh, (long)BSZ * DM);

  // act1 (gelu) -> act2 -> top-128 -> sigmoid scatter
  gemm<2, true, f16>(st, pooledh, act1w, FP(33), a1h, BSZ, 512, DM);
  gemm<0, true, float>(st, a1h, act2w, FP(35), logits0, BSZ, NN, 512);
  topk_kernel<<<BSZ, 256, 0, st>>>(logits0, NN, 128, idx, vals);
  zfill(st, acta, (long)BSZ * NN);
  scatter_vals_kernel<<<gblocks((long)BSZ * 128), 256, 0, st>>>(idx, vals, acta, NN, 128, 1);

  // ================= interaction steps =================
  init_hidden_kernel<<<gblocks(BND), 256, 0, st>>>(FP(36), hid0);
  float* hcur = hid0;
  float* hnext = hid1;

  for (int step = 0; step < 5; ++step) {
    const int k = (step == 0) ? 128 : 256;
    const long BK_ = (long)BSZ * k;

    // --- compute_messages: gather -> MHA -> weighted scatter ---
    topk_kernel<<<BSZ, 256, 0, st>>>(acta, NN, k, idx, vals);
    gather_rows_kernel<<<gblocks(BK_ * DM), 256, 0, st>>>(hcur, idx, activeh, k);

    gemm<0, true, f16>(st, activeh, ma_wi, FP(38), qkvh, (int)BK_, 3 * DM, DM);
    split_qkv_kernel<<<gblocks(BK_ * 3 * DM), 256, 0, st>>>(qkvh, Qh, Kh, Vh, BSZ, k);
    gemm<0, true, float>(st, Qh, Kh, nullptr, scores, k, k, HDIM,
                         BSZ * NHD, (long)k * HDIM, (long)k * HDIM, (long)k * k, 0.125f);
    softmax_kernel<<<rblocks((long)BSZ * NHD * k), 256, 0, st>>>(
        scores, probs, (long)BSZ * NHD * k, k);
    gemm<0, false, f16>(st, probs, Vh, nullptr, Oh, k, HDIM, k,
                        BSZ * NHD, (long)k * k, (long)k * HDIM, (long)k * HDIM, 1.f);
    merge_heads_kernel<<<gblocks(BK_ * DM), 256, 0, st>>>(Oh, ctx, BSZ, k);
    gemm<0, true, float>(st, ctx, ma_wo, FP(40), msgout, (int)BK_, DM, DM);

    build_comb_kernel<<<gblocks(BND), 256, 0, st>>>(hcur, comb);
    scatter_msg_kernel<<<gblocks(BK_ * DM), 256, 0, st>>>(comb, idx, vals, msgout, k);

    // --- update_states: su1(gelu) -> su2 -> LN ---
    gemm<2, true, f16>(st, comb, su1w, FP(42), t1h, (int)((long)BSZ * NN), DM, 2 * DM);
    gemm<0, true, float>(st, t1h, su2w, FP(44), nh, (int)((long)BSZ * NN), DM, DM);
    add_ln_kernel<<<rblocks((long)BSZ * NN), 256, 0, st>>>(
        nh, nullptr, FP(45), FP(46), nh, nhh, (long)BSZ * NN);

    // --- update_activations: au1(gelu) -> au2(sigmoid) ---
    build_comb2_kernel<<<gblocks(BND), 256, 0, st>>>(hcur, nhh, comb);
    gemm<2, true, f16>(st, comb, au1w, FP(48), t1h, (int)((long)BSZ * NN), DM, 2 * DM);
    rowdot_sigmoid_kernel<<<rblocks((long)BSZ * NN), 256, 0, st>>>(
        t1h, FP(49), FP(50), delta, (long)BSZ * NN);
    update_act_kernel<<<gblocks((long)BSZ * NN), 256, 0, st>>>(acta, delta, actb);

    // --- maintain_sparsity: top-256 ---
    topk_kernel<<<BSZ, 256, 0, st>>>(actb, NN, 256, idx, vals);
    zfill(st, acta, (long)BSZ * NN);
    scatter_vals_kernel<<<gblocks((long)BSZ * 256), 256, 0, st>>>(idx, vals, acta, NN, 256, 0);
    zfill(st, hnext, BND);
    scatter_rows_kernel<<<gblocks((long)BSZ * 256 * DM), 256, 0, st>>>(nh, idx, hnext, 256);

    float* t = hcur; hcur = hnext; hnext = t;
  }

  // ================= output decoder =================
  cvt(st, acta, acth, (long)BSZ * NN);
  gemm<2, true, f16>(st, acth, dec1w, FP(52), d1h, BSZ, 512, NN);
  gemm<0, true, float>(st, d1h, dec2w, FP(54), d2, BSZ, DM, 512);
  add_ln_kernel<<<rblocks(BSZ), 256, 0, st>>>(d2, nullptr, FP(55), FP(56), nullptr, dnh, BSZ);
  gemm<0, true, float>(st, dnh, logw, FP(58), (float*)d_out, BSZ, VOC, DM);
}
